// CrossModalSCA_76312978915551
// MI455X (gfx1250) — matrix operand order repaired and checked by reading, not compiled, not run on hardware
//
#include <hip/hip_runtime.h>
#include <cstddef>
#include <cstdint>

// ---------------- problem constants (fixed by harness) ----------------
constexpr int kNQ    = 10000;
constexpr int kCams  = 6;
constexpr int kHeads = 8;
constexpr int kLTot  = 30825;           // sum of H*W over levels
constexpr int kVRows = kCams * kLTot;   // 184950 rows of value

__constant__ int g_lvlW[4]    = {200, 100, 50, 25};
__constant__ int g_lvlH[4]    = {116, 58, 29, 15};
__constant__ int g_lvlBase[4] = {0, 23200, 29000, 30450};

// ---------------- vector types ----------------
typedef __attribute__((ext_vector_type(16))) __bf16 v16bf;
typedef __attribute__((ext_vector_type(8)))  __bf16 v8bf;
typedef __attribute__((ext_vector_type(2)))  __bf16 v2bf;
typedef __attribute__((ext_vector_type(8)))  float  v8f;

// ---------------- WMMA GEMM: C[M x N] = A[M x 128] @ W[128 x N] + bias ----------------
// Block tile: 64 rows x 128 cols, 256 threads (8 waves). Each wave owns one
// 16-col tile and iterates 4 M-subtiles, with its 4 B fragments held in
// registers (loaded once from a transposed bf16 W image in LDS).
// Fragment loads are two aligned 16B LDS reads each (ds_load_b128),
// bank-conflict-free with the 136-element padded stride.
// N / relu / residual / output-type are compile-time -> shift addressing,
// immediate store offsets, no per-element flag branches.
// EXEC is all-ones through every v_wmma (tail guards only in the last block).
template <int N, bool RELU, bool RES, bool OUTBF>
__global__ void gemm128_wmma(const float* __restrict__ A,
                             const float* __restrict__ W,
                             const float* __restrict__ bias,
                             const float* __restrict__ residual,
                             float* __restrict__ Cf,
                             __bf16* __restrict__ Cbf,
                             int M)
{
    constexpr int LT = 136;                 // padded stride: 272B rows -> 16B-aligned runs
    __shared__ __bf16 ldsA[64 * LT];        // A tile, row-major           (17408 B)
    __shared__ __bf16 ldsW[128 * LT];       // W^T tile: [col][k]          (34816 B)

    const int mb0  = blockIdx.x * 64;
    const int nb0  = blockIdx.y * 128;
    const int tid  = threadIdx.x;
    const int wave = tid >> 5;
    const int lane = tid & 31;
    const int ml   = lane & 15;
    const int hi   = lane >> 4;
    const int colT = wave * 16 + ml;        // column within the 128-wide tile
    const bool fullTile = (mb0 + 64 <= M);

    // ---- stage A (64x128) -> bf16 LDS, one float2 per thread-iteration ----
    for (int i = tid; i < 64 * 64; i += 256) {
        int r  = i >> 6;                    // row 0..63
        int c  = (i & 63) * 2;              // col 0..126 step 2
        int row = mb0 + r;
        float2 v = (fullTile || row < M) ? *(const float2*)(A + (size_t)row * 128 + c)
                                         : make_float2(0.0f, 0.0f);
        v2bf pk; pk[0] = (__bf16)v.x; pk[1] = (__bf16)v.y;
        *(v2bf*)(ldsA + r * LT + c) = pk;
    }
    // ---- stage W^T (128k x 128col): coalesced global float2, transposed write ----
    for (int i = tid; i < 128 * 64; i += 256) {
        int k = i >> 6;
        int c = (i & 63) * 2;
        float2 v = *(const float2*)(W + (size_t)k * N + nb0 + c);
        ldsW[(c    ) * LT + k] = (__bf16)v.x;
        ldsW[(c + 1) * LT + k] = (__bf16)v.y;
    }
    __syncthreads();

    // ---- preload this wave's 4 B fragments (held across all M-subtiles) ----
    // B 32x16 bf16 layout: lane (hi,ml) holds K = kc*32 + 16*hi + (0..15), col = ml.
    v16bf bfrag[4];
    #pragma unroll
    for (int kc = 0; kc < 4; ++kc) {
        const __bf16* bp = ldsW + colT * LT + kc * 32 + hi * 16;
        v8bf b0 = *(const v8bf*)bp;
        v8bf b1 = *(const v8bf*)(bp + 8);
        bfrag[kc] = __builtin_shufflevector(b0, b1,
                        0,1,2,3,4,5,6,7,8,9,10,11,12,13,14,15);
    }

    const int colg = nb0 + colT;
    const float bsv = bias[colg];

    #pragma unroll
    for (int ms = 0; ms < 4; ++ms) {
        v8f acc = {};
        #pragma unroll
        for (int kc = 0; kc < 4; ++kc) {
            // A 16x32 bf16 layout: lane (hi,ml) holds one contiguous K-run:
            // [kc*32 + 16*hi, +16), row = ml.
            const __bf16* ap = ldsA + (ms * 16 + ml) * LT + kc * 32 + hi * 16;
            v8bf a0 = *(const v8bf*)ap;
            v8bf a1 = *(const v8bf*)(ap + 8);
            v16bf afr = __builtin_shufflevector(a0, a1,
                            0,1,2,3,4,5,6,7,8,9,10,11,12,13,14,15);
            acc = __builtin_amdgcn_wmma_f32_16x16x32_bf16(
                      false, afr, false, bfrag[kc], (short)0, acc, false, false);
        }
        // C/D layout: VGPR r -> row = row0 + r (+8 for hi half), col = colg.
        // Base pointers hoisted; r*N is a compile-time immediate offset.
        const int row0 = mb0 + ms * 16 + hi * 8;
        const size_t base = (size_t)((unsigned)row0 * (unsigned)N) + (unsigned)colg;
        if (fullTile) {
            #pragma unroll
            for (int r = 0; r < 8; ++r) {
                float v = acc[r] + bsv;
                if constexpr (RELU) v = fmaxf(v, 0.0f);
                if constexpr (RES)  v += residual[base + (size_t)r * N];
                if constexpr (OUTBF) Cbf[base + (size_t)r * N] = (__bf16)v;
                else                 Cf [base + (size_t)r * N] = v;
            }
        } else {
            #pragma unroll
            for (int r = 0; r < 8; ++r) {
                if (row0 + r < M) {
                    float v = acc[r] + bsv;
                    if constexpr (RELU) v = fmaxf(v, 0.0f);
                    if constexpr (RES)  v += residual[base + (size_t)r * N];
                    if constexpr (OUTBF) Cbf[base + (size_t)r * N] = (__bf16)v;
                    else                 Cf [base + (size_t)r * N] = v;
                }
            }
        }
    }
}

// ---------------- LayerNorm over last dim (128), one block per row ----------------
__global__ void layernorm128(const float* __restrict__ x,
                             const float* __restrict__ g,
                             const float* __restrict__ b,
                             float* __restrict__ y, int rows)
{
    int row = blockIdx.x;
    int tid = threadIdx.x;           // 0..127
    int lane = tid & 31, wave = tid >> 5;
    float v = x[(size_t)row * 128 + tid];
    float s = v, s2 = v * v;
    #pragma unroll
    for (int off = 16; off; off >>= 1) {
        s  += __shfl_xor(s,  off, 32);
        s2 += __shfl_xor(s2, off, 32);
    }
    __shared__ float rs[4], rs2[4];
    if (lane == 0) { rs[wave] = s; rs2[wave] = s2; }
    __syncthreads();
    float S  = rs[0]  + rs[1]  + rs[2]  + rs[3];
    float S2 = rs2[0] + rs2[1] + rs2[2] + rs2[3];
    float mean = S * (1.0f / 128.0f);
    float var  = S2 * (1.0f / 128.0f) - mean * mean;
    float inv  = rsqrtf(var + 1e-5f);
    y[(size_t)row * 128 + tid] = (v - mean) * inv * g[tid] + b[tid];
}

// ---------------- softmax over 32 (LEVELS*POINTS) per (q, head), in-place ----------------
// block = 256 (8 waves = 8 heads), grid = NQ
__global__ void softmax32(float* __restrict__ aw)
{
    int q = blockIdx.x;
    int h = threadIdx.x >> 5;
    int lane = threadIdx.x & 31;
    size_t idx = (size_t)q * 256 + h * 32 + lane;
    float v = aw[idx];
    float m = v;
    #pragma unroll
    for (int off = 16; off; off >>= 1) m = fmaxf(m, __shfl_xor(m, off, 32));
    float e = __expf(v - m);
    float s = e;
    #pragma unroll
    for (int off = 16; off; off >>= 1) s += __shfl_xor(s, off, 32);
    aw[idx] = e / s;
}

// ---------------- MSDA bilinear sampler ----------------
// One wave per (cam, q). Half-waves own one sample each; lanes 0-15 own the
// 16 contiguous channels of the current head -> 32B bursts per corner (bf16,
// L2-resident). grid = (NQ/8, CAMS), block = 256.
__global__ void msda_sample(const __bf16* __restrict__ value,   // [CAMS*LTOT, 128] bf16
                            const float* __restrict__ so,       // [NQ, 8,4,8,2] raw offsets
                            const float* __restrict__ aw,       // [NQ, 8,32] softmaxed
                            const float* __restrict__ refpts,   // [CAMS, NQ, 4, 2]
                            const unsigned int* __restrict__ mask4, // [CAMS*NQ] 4 packed bools
                            float* __restrict__ out_cam)        // [CAMS, NQ, 128]
{
    const int cam  = blockIdx.y;
    const int q    = blockIdx.x * 8 + (threadIdx.x >> 5);
    const int lane = threadIdx.x & 31;
    const int d    = lane & 15;
    const int hiw  = lane >> 4;

    float* outp = out_cam + ((size_t)cam * kNQ + q) * 128;
    bool valid = mask4[(size_t)cam * kNQ + q] != 0u;
    if (!valid) {
        for (int i = lane; i < 128; i += 32) outp[i] = 0.0f;
        return;
    }

    const float* ref = refpts + ((size_t)cam * kNQ + q) * 8;   // 4 z-anchors x 2
    const int camBase = cam * kLTot;

    for (int h = 0; h < kHeads; ++h) {
        float accv = 0.0f;
        #pragma unroll 4
        for (int it = 0; it < 16; ++it) {
            int s = it * 2 + hiw;          // sample 0..31 = l*8 + p
            int l = s >> 3;
            int p = s & 7;
            int z = p & 3;                 // (ppa, D_Z) flatten: z = p % 4
            int Wl = g_lvlW[l], Hl = g_lvlH[l], base = g_lvlBase[l];

            size_t soff = (size_t)q * 512 + h * 64 + l * 16 + p * 2;
            float wgt = aw[(size_t)q * 256 + h * 32 + s];

            // (ref + so/W)*W - 0.5 == ref*W + so - 0.5  (division folded away)
            float x = ref[z * 2 + 0] * (float)Wl + so[soff + 0] - 0.5f;
            float y = ref[z * 2 + 1] * (float)Hl + so[soff + 1] - 0.5f;
            float x0 = floorf(x), y0 = floorf(y);
            float wx = x - x0, wy = y - y0;
            int xi = (int)x0, yi = (int)y0;

            auto corner = [&](int cx, int cy) -> float {
                bool ok = (cx >= 0) && (cx < Wl) && (cy >= 0) && (cy < Hl);
                int xc = min(max(cx, 0), Wl - 1);
                int yc = min(max(cy, 0), Hl - 1);
                float v = (float)value[(size_t)(camBase + base + yc * Wl + xc) * 128 + h * 16 + d];
                return ok ? v : 0.0f;
            };
            float smp = corner(xi,     yi    ) * (1.0f - wx) * (1.0f - wy)
                      + corner(xi + 1, yi    ) * wx          * (1.0f - wy)
                      + corner(xi,     yi + 1) * (1.0f - wx) * wy
                      + corner(xi + 1, yi + 1) * wx          * wy;
            accv += wgt * smp;
        }
        accv += __shfl_xor(accv, 16, 32);   // combine the two half-wave partials
        if (hiw == 0) outp[h * 16 + d] = accv;
    }
}

// ---------------- masked mean over cameras ----------------
// grid = NQ, block = 128
__global__ void reduce_cams(const float* __restrict__ out_cam,
                            const unsigned int* __restrict__ mask4,
                            float* __restrict__ slots)
{
    int q = blockIdx.x;
    int d = threadIdx.x;
    float s = 0.0f, cnt = 0.0f;
    #pragma unroll
    for (int cam = 0; cam < kCams; ++cam) {
        cnt += (mask4[(size_t)cam * kNQ + q] != 0u) ? 1.0f : 0.0f;
        s += out_cam[((size_t)cam * kNQ + q) * 128 + d];  // masked cams wrote zeros
    }
    slots[(size_t)q * 128 + d] = s / fmaxf(cnt, 1.0f);
}

// ---------------- host-side orchestration ----------------
extern "C" void kernel_launch(void* const* d_in, const int* in_sizes, int n_in,
                              void* d_out, int out_size, void* d_ws, size_t ws_size,
                              hipStream_t stream) {
    (void)in_sizes; (void)n_in; (void)out_size; (void)ws_size;

    const float* query  = (const float*)d_in[0];   // (1, NQ, 128)
    const float* feats  = (const float*)d_in[1];   // (6, LTOT, 1, 128) == value rows
    const float* refpts = (const float*)d_in[2];   // (6, 1, NQ, 4, 2)
    const unsigned int* mask4 = (const unsigned int*)d_in[3]; // bool (6,1,NQ,4) -> u32/query
    // d_in[4] spatial_shapes, d_in[5] level_start_index: compile-time constants here
    const float* n1g  = (const float*)d_in[6];
    const float* n1b  = (const float*)d_in[7];
    const float* so_w = (const float*)d_in[8];
    const float* so_b = (const float*)d_in[9];
    const float* aw_w = (const float*)d_in[10];
    const float* aw_b = (const float*)d_in[11];
    const float* vp_w = (const float*)d_in[12];
    const float* vp_b = (const float*)d_in[13];
    const float* op_w = (const float*)d_in[14];
    const float* op_b = (const float*)d_in[15];
    const float* n2g  = (const float*)d_in[16];
    const float* n2b  = (const float*)d_in[17];
    const float* f1_w = (const float*)d_in[18];
    const float* f1_b = (const float*)d_in[19];
    const float* f2_w = (const float*)d_in[20];
    const float* f2_b = (const float*)d_in[21];

    // workspace carve-up (~135 MB total)
    char* ws = (char*)d_ws;
    size_t off = 0;
    auto carve = [&](size_t bytes) -> void* {
        void* p = ws + off;
        off = (off + bytes + 255) & ~(size_t)255;
        return p;
    };
    float*  qn      = (float*) carve((size_t)kNQ * 128 * 4);        // LN1(query)
    __bf16* valbf   = (__bf16*)carve((size_t)kVRows * 128 * 2);     // projected value, bf16
    float*  so      = (float*) carve((size_t)kNQ * 512 * 4);        // raw sampling offsets
    float*  aw      = (float*) carve((size_t)kNQ * 256 * 4);        // attn weights
    float*  out_cam = (float*) carve((size_t)kCams * kNQ * 128 * 4);
    float*  slots   = (float*) carve((size_t)kNQ * 128 * 4);
    float*  fused   = (float*) carve((size_t)kNQ * 128 * 4);
    float*  normed  = (float*) carve((size_t)kNQ * 128 * 4);
    float*  h1      = (float*) carve((size_t)kNQ * 128 * 4);

    const int gQ = (kNQ + 63) / 64;          // 157 row-blocks for NQ-sized GEMMs
    const int gV = (kVRows + 63) / 64;       // 2890 row-blocks for value GEMM

    // 1) q = LN1(query)
    layernorm128<<<kNQ, 128, 0, stream>>>(query, n1g, n1b, qn, kNQ);

    // 2) value = feats @ vp_w + vp_b  (184950 x 128, bf16 out)  -- biggest GEMM
    gemm128_wmma<128, false, false, true><<<dim3(gV, 1), 256, 0, stream>>>(
        feats, vp_w, vp_b, nullptr, nullptr, valbf, kVRows);

    // 3) so/aw from q ONCE (camera-broadcast exploited: 6x fewer rows than ref)
    gemm128_wmma<512, false, false, false><<<dim3(gQ, 4), 256, 0, stream>>>(
        qn, so_w, so_b, nullptr, so, nullptr, kNQ);
    gemm128_wmma<256, false, false, false><<<dim3(gQ, 2), 256, 0, stream>>>(
        qn, aw_w, aw_b, nullptr, aw, nullptr, kNQ);
    softmax32<<<kNQ, 256, 0, stream>>>(aw);

    // 4) deformable bilinear sampling per camera (L2-resident bf16 value)
    msda_sample<<<dim3(kNQ / 8, kCams), 256, 0, stream>>>(
        valbf, so, aw, refpts, mask4, out_cam);

    // 5) masked mean over cameras
    reduce_cams<<<kNQ, 128, 0, stream>>>(out_cam, mask4, slots);

    // 6) fused = slots @ op_w + op_b + q
    gemm128_wmma<128, false, true, false><<<dim3(gQ, 1), 256, 0, stream>>>(
        slots, op_w, op_b, qn, fused, nullptr, kNQ);

    // 7) normed = LN2(fused)
    layernorm128<<<kNQ, 128, 0, stream>>>(fused, n2g, n2b, normed, kNQ);

    // 8) FFN: out = normed + relu(normed @ f1 + b1) @ f2 + b2
    gemm128_wmma<128, true, false, false><<<dim3(gQ, 1), 256, 0, stream>>>(
        normed, f1_w, f1_b, nullptr, h1, nullptr, kNQ);
    gemm128_wmma<128, false, true, false><<<dim3(gQ, 1), 256, 0, stream>>>(
        h1, f2_w, f2_b, normed, (float*)d_out, nullptr, kNQ);
}
